// MultiheadAttention_67662914781836
// MI455X (gfx1250) — matrix operand order repaired and checked by reading, not compiled, not run on hardware
//
#include <hip/hip_runtime.h>
#include <hip/hip_bf16.h>

typedef __attribute__((ext_vector_type(16))) _Float16 v16h;
typedef __attribute__((ext_vector_type(8)))  _Float16 v8h;
typedef __attribute__((ext_vector_type(4)))  _Float16 v4h;
typedef __attribute__((ext_vector_type(8)))  float    v8f;
typedef __attribute__((ext_vector_type(4)))  float    v4f;

#define SEQ    2048
#define BATCH  4
#define NHEAD  8
#define HEADD  32      // qk head dim == v head dim
#define DIMQ   256
#define DIMV   256

// log2-domain constants: scores computed as (q*scale*log2e)·k so exp() -> v_exp_f32 (exp2)
#define QSCALE (0.17677669529663687f * 1.4426950408889634f)   // (1/sqrt(32)) * log2(e)
#define BIASC  (-1.4426950408889634e30f)                      // -1e30 * log2(e)

union H16 { v16h v; v8h h2[2]; _Float16 h[16]; unsigned u[8]; };

static __device__ inline v8f wmma_f16(const v16h& a, const v16h& b, const v8f& c) {
  return __builtin_amdgcn_wmma_f32_16x16x32_f16(false, a, false, b, (short)0, c, false, false);
}

// ---- pre-pass: additive mask bias in log2 domain, computed once into d_ws (32 KB)
__global__ void build_bias(const float* __restrict__ mask, float* __restrict__ bias) {
  const int i = blockIdx.x * 256 + threadIdx.x;
  bias[i] = (1.0f - mask[i]) * BIASC;
}

// Flash-attention forward. One workgroup = one (b,h), 128 query rows (8 waves x 16).
// GEMM1 computes S^T = K * Q^T (softmax state lane-local up to one shfl_xor(16));
// exp'd scores feed GEMM2's B operand (out^T = V^T * P^T) directly.
// K and V chunks are converted f32->f16 cooperatively once per WG and staged in LDS
// (K row-major for A-operand b128 loads, V transposed), double buffered, 1 barrier/iter.
__launch_bounds__(256, 2)
__global__ void mha_flash_wmma(const float* __restrict__ q,
                               const float* __restrict__ k,
                               const float* __restrict__ v,
                               const float* __restrict__ bias,
                               float* __restrict__ out) {
  constexpr int RS = 40;                            // LDS row stride in halfs (80B: 16B-aligned, bank-spread)
  __shared__ __align__(16) _Float16 KT[2][32 * RS]; // K tile, row-major  [key][dim]
  __shared__ __align__(16) _Float16 VT[2][32 * RS]; // V tile, transposed [dv][key]

  const int b      = blockIdx.y >> 3;               // / NHEAD
  const int h      = blockIdx.y & 7;
  const int lane   = threadIdx.x & 31;
  const int lane15 = lane & 15;
  const int hi     = lane >> 4;                     // 0 = low half, 1 = high half
  const int q0     = blockIdx.x * 128 + (threadIdx.x >> 5) * 16;

  // staging roles (per thread): one 32-key chunk = 32 keys x 32 dims
  const int key_in = threadIdx.x >> 3;              // 0..31
  const int dg     = threadIdx.x & 7;               // 0..7 (4 dims each)
  const float* kstage = k + (size_t)b * DIMQ + h * HEADD + dg * 4;   // + key*BATCH*DIMQ
  const float* vstage = v + (size_t)b * DIMV + h * HEADD + dg * 4;

  // ---- Q^T as B-operand of GEMM1 (loaded once): lane column = query, k-idx e -> dim e + 16*hi
  H16 bq;
  {
    const float* qp = q + ((size_t)(q0 + lane15) * BATCH + b) * DIMQ + h * HEADD + hi * 16;
#pragma unroll
    for (int j = 0; j < 4; ++j) {
      v4f t = *(const v4f*)(qp + 4 * j);
#pragma unroll
      for (int e = 0; e < 4; ++e) bq.h[4 * j + e] = (_Float16)(t[e] * QSCALE);
    }
  }

  v8f acc0 = {}, acc1 = {};                         // out^T accumulators: dv 0..15, 16..31
  float mrun = -INFINITY, lsum = 0.0f;
  const float* bbase = bias + (size_t)b * SEQ;

  auto stage = [&](int kc, int buf) {
    const size_t krow = (size_t)(kc + key_in) * (BATCH * DIMQ);
    const size_t vrow = (size_t)(kc + key_in) * (BATCH * DIMV);
    v4f tk = *(const v4f*)(kstage + krow);
    v4f tv = *(const v4f*)(vstage + vrow);
    // K: row-major; element-wise f16 converts so clang emits v_cvt_pk_f16_f32, then one b64 store
    v4h hk;
#pragma unroll
    for (int j = 0; j < 4; ++j) hk[j] = (_Float16)tk[j];
    *(v4h*)(&KT[buf][key_in * RS + dg * 4]) = hk;
    // V: transposed scatter (4 x b16)
#pragma unroll
    for (int j = 0; j < 4; ++j)
      VT[buf][(dg * 4 + j) * RS + key_in] = (_Float16)tv[j];
    // prefetch two chunks ahead
    if (kc + 64 < SEQ) {
      __builtin_prefetch(kstage + krow + (size_t)64 * BATCH * DIMQ, 0, 0);
      __builtin_prefetch(vstage + vrow + (size_t)64 * BATCH * DIMV, 0, 0);
    }
  };

  stage(0, 0);
  __syncthreads();

  for (int kc = 0; kc < SEQ; kc += 32) {
    const int buf = (kc >> 5) & 1;
    if (kc + 32 < SEQ) stage(kc + 32, buf ^ 1);

    const _Float16* kt = KT[buf];
    const _Float16* vt = VT[buf];
    const int g = hi * 16;

    // ---- GEMM1: S^T tiles from LDS. A rows = keys; e -> dim g+e (contiguous).
    H16 ak0, ak1;
    ak0.h2[0] = *(const v8h*)(kt + lane15 * RS + g);
    ak0.h2[1] = *(const v8h*)(kt + lane15 * RS + g + 8);
    ak1.h2[0] = *(const v8h*)(kt + (16 + lane15) * RS + g);
    ak1.h2[1] = *(const v8h*)(kt + (16 + lane15) * RS + g + 8);
    v8f s0 = {}, s1 = {};
    s0 = wmma_f16(ak0.v, bq.v, s0);
    s1 = wmma_f16(ak1.v, bq.v, s1);

    // ---- additive mask bias (precomputed): vector adds -> v_pk_add_f32
    v8f b0v, b1v;
    {
      const float* bp_ = bbase + kc + hi * 8;
      v4f b0a = *(const v4f*)(bp_);
      v4f b0b = *(const v4f*)(bp_ + 4);
      v4f b1a = *(const v4f*)(bp_ + 16);
      v4f b1b = *(const v4f*)(bp_ + 20);
#pragma unroll
      for (int r = 0; r < 4; ++r) {
        b0v[r] = b0a[r]; b0v[r + 4] = b0b[r];
        b1v[r] = b1a[r]; b1v[r + 4] = b1b[r];
      }
    }
    const v8f st0v = s0 + b0v;
    const v8f st1v = s1 + b1v;

    // ---- online softmax in log2 domain (one shfl_xor(16) per reduction)
    float mloc = fmaxf(st0v[0], st1v[0]);
#pragma unroll
    for (int r = 1; r < 8; ++r) mloc = fmaxf(mloc, fmaxf(st0v[r], st1v[r]));
    mloc = fmaxf(mloc, __shfl_xor(mloc, 16, 32));
    const float mnew = fmaxf(mrun, mloc);

    if (__any((int)(mnew > mrun))) {                // rescale only when some max moved
      const float alpha = __builtin_amdgcn_exp2f(mrun - mnew);
      acc0 *= alpha;                                // v_pk_mul_f32
      acc1 *= alpha;
      lsum *= alpha;
    }
    mrun = mnew;

    const v8f d0 = st0v - mnew;                     // v_pk_add_f32 (neg)
    const v8f d1 = st1v - mnew;
    v8f p0v, p1v;
#pragma unroll
    for (int r = 0; r < 8; ++r) {
      p0v[r] = __builtin_amdgcn_exp2f(d0[r]);       // v_exp_f32
      p1v[r] = __builtin_amdgcn_exp2f(d1[r]);
    }
    const v8f ps = p0v + p1v;                       // v_pk_add_f32
    float slocal = ((ps[0] + ps[1]) + (ps[2] + ps[3])) + ((ps[4] + ps[5]) + (ps[6] + ps[7]));
    lsum += slocal + __shfl_xor(slocal, 16, 32);

    // ---- pack P^T (element-wise converts -> v_cvt_pk_f16_f32), one cross-half exchange
    H16 P;
#pragma unroll
    for (int r = 0; r < 8; ++r) {
      P.h[r]     = (_Float16)p0v[r];
      P.h[8 + r] = (_Float16)p1v[r];
    }
    H16 bp;
#pragma unroll
    for (int j = 0; j < 4; ++j) {
      const unsigned R0 = __shfl_xor(P.u[j], 16, 32);       // partner's p0 words
      const unsigned R1 = __shfl_xor(P.u[4 + j], 16, 32);   // partner's p1 words
      bp.u[j]     = hi ? R1 : P.u[j];
      bp.u[4 + j] = hi ? P.u[4 + j] : R0;
    }

    // ---- GEMM2: out^T += V^T * P^T (A rows = dv, from transposed LDS tile)
    H16 av0, av1;
    av0.h2[0] = *(const v8h*)(vt + lane15 * RS + g);
    av0.h2[1] = *(const v8h*)(vt + lane15 * RS + g + 8);
    av1.h2[0] = *(const v8h*)(vt + (16 + lane15) * RS + g);
    av1.h2[1] = *(const v8h*)(vt + (16 + lane15) * RS + g + 8);
    acc0 = wmma_f16(av0.v, bp.v, acc0);
    acc1 = wmma_f16(av1.v, bp.v, acc1);

    __syncthreads();
  }

  // ---- epilogue: out^T D-layout -> (S,B,Dv). lane<16 owns dv {0..7,16..23}, lane>=16 {8..15,24..31}.
  const float inv = (lsum > 0.0f) ? __builtin_amdgcn_rcpf(lsum) : 0.0f;
  const v8f o0v = acc0 * inv;
  const v8f o1v = acc1 * inv;
  float* op = out + ((size_t)(q0 + lane15) * BATCH + b) * DIMV + h * HEADD + hi * 8;
  v4f o0, o1, o2, o3;
#pragma unroll
  for (int e = 0; e < 4; ++e) {
    o0[e] = o0v[e];
    o1[e] = o0v[4 + e];
    o2[e] = o1v[e];
    o3[e] = o1v[4 + e];
  }
  *(v4f*)(op)      = o0;
  *(v4f*)(op + 4)  = o1;
  *(v4f*)(op + 16) = o2;
  *(v4f*)(op + 20) = o3;
}

extern "C" void kernel_launch(void* const* d_in, const int* in_sizes, int n_in,
                              void* d_out, int out_size, void* d_ws, size_t ws_size,
                              hipStream_t stream) {
  (void)in_sizes; (void)n_in; (void)ws_size; (void)out_size;
  const float* q    = (const float*)d_in[0];
  const float* k    = (const float*)d_in[1];
  const float* v    = (const float*)d_in[2];
  const float* mask = (const float*)d_in[3];
  float* bias       = (float*)d_ws;                 // BATCH*SEQ floats = 32 KB
  float* out        = (float*)d_out;

  build_bias<<<(BATCH * SEQ) / 256, 256, 0, stream>>>(mask, bias);

  dim3 grid(SEQ / 128, BATCH * NHEAD);              // 16 x 32 workgroups
  dim3 block(256);                                  // 8 waves (wave32)
  mha_flash_wmma<<<grid, block, 0, stream>>>(q, k, v, bias, out);
}
